// MS_ResBlock_7782480740670
// MI455X (gfx1250) — matrix operand order repaired and checked by reading, not compiled, not run on hardware
//
#include <hip/hip_runtime.h>
#include <cstdint>
#include <cstddef>

// ---------------------------------------------------------------------------
// MS_ResBlock: Y = LN(X @ W^T + b); spikes = LIF_scan(Y + X)
// X: [T=32, B=128, D=2048] f32; W: [D,D] f32 row-major; out: spikes f32.
// GEMM: M=4096 (T*B), N=K=2048.  bf16 WMMA, f32 accumulate.
// Fast path: X,W preconverted to bf16 in d_ws; tiles staged with
// global_load_async_to_lds_b128 (ASYNCcnt), double-buffered; zero VALU in
// the WMMA loop.  Fallback (small d_ws): sync staging with pack-at-stage.
// ---------------------------------------------------------------------------

typedef __attribute__((ext_vector_type(16))) __bf16       v16bf;
typedef __attribute__((ext_vector_type(8)))  float        v8f;
typedef __attribute__((ext_vector_type(4)))  unsigned int v4u;
typedef __attribute__((ext_vector_type(4)))  float        v4f;
typedef __attribute__((ext_vector_type(4)))  int          v4i;

constexpr int T_STEPS = 32;
constexpr int BATCH   = 128;
constexpr int DIM     = 2048;            // N and K
constexpr int MTOT    = T_STEPS * BATCH; // 4096
constexpr int BM = 128, BN = 128, BK = 32;
constexpr int LDH = 40;                  // bf16 per LDS row: 32 data + 8 pad (80B, 16B-aligned)

#if defined(__gfx1250__) && __has_builtin(__builtin_amdgcn_global_load_async_to_lds_b128)
#define HAVE_ASYNC 1
#else
#define HAVE_ASYNC 0
#endif

__device__ __forceinline__ unsigned pack_bf16x2(float a, float b) {
  unsigned ua = __float_as_uint(a);
  unsigned ub = __float_as_uint(b);
  ua = ua + 0x7FFFu + ((ua >> 16) & 1u);   // RNE f32 -> bf16
  ub = ub + 0x7FFFu + ((ub >> 16) & 1u);
  return (ua >> 16) | (ub & 0xFFFF0000u);
}

__device__ __forceinline__ v4u pack8(v4f a, v4f b) {
  v4u r;
  r.x = pack_bf16x2(a.x, a.y);
  r.y = pack_bf16x2(a.z, a.w);
  r.z = pack_bf16x2(b.x, b.y);
  r.w = pack_bf16x2(b.z, b.w);
  return r;
}

#if HAVE_ASYNC
typedef __attribute__((address_space(1))) v4i* gbl_v4i_p;   // global (AS1)
typedef __attribute__((address_space(3))) v4i* lds_v4i_p;   // LDS (AS3)
#endif

// 16B global -> LDS copy: async engine if available, else load+store.
__device__ __forceinline__ void cp_b128(const void* g, void* l) {
#if HAVE_ASYNC
  __builtin_amdgcn_global_load_async_to_lds_b128((gbl_v4i_p)g, (lds_v4i_p)l, 0, 0);
#else
  *(v4u*)l = *(const v4u*)g;
#endif
}

template <int N>
__device__ __forceinline__ void wait_async() {
#if HAVE_ASYNC
# if __has_builtin(__builtin_amdgcn_s_wait_asynccnt)
  __builtin_amdgcn_s_wait_asynccnt(N);
# else
  asm volatile("s_wait_asynccnt %0" :: "n"(N) : "memory");
# endif
#endif
}

union Frag { v4u q[2]; v16bf v; };

// ---------------------------------------------------------------------------
// Kernel 0 (fast path): elementwise f32 -> bf16, 8 elems/thread.
// ---------------------------------------------------------------------------
__global__ __launch_bounds__(256)
void f32_to_bf16_kernel(const float* __restrict__ src,
                        unsigned short* __restrict__ dst, int n8) {
  int i = blockIdx.x * 256 + threadIdx.x;
  if (i >= n8) return;
  const v4f* s = (const v4f*)src + (size_t)i * 2;
  *(v4u*)(dst + (size_t)i * 8) = pack8(s[0], s[1]);
}

// ---------------------------------------------------------------------------
// Kernel 1 (fast path): Y[m,n] = sum_k Xb[m,k]*Wb[n,k] + bias[n]  (f32 out)
// bf16 operands in global; async->LDS staging, double-buffered; no VALU in
// the WMMA loop.  Block 256 (8 waves), tile 128x128, K-step 32.
// Wave grid 4(M) x 2(N); each wave: 2x4 grid of 16x16 WMMA tiles.
// ---------------------------------------------------------------------------
__global__ __launch_bounds__(256)
void gemm_bias_bf16_kernel(const unsigned short* __restrict__ Xb,
                           const unsigned short* __restrict__ Wb,
                           const float* __restrict__ bias,
                           float* __restrict__ Y) {
  __shared__ unsigned short lsA[2][BM * LDH];   // 10KB each buffer
  __shared__ unsigned short lsB[2][BN * LDH];

  const int tid  = threadIdx.x;
  const int lane = tid & 31;
  const int wid  = tid >> 5;
  const int wm   = wid & 3;
  const int wn   = wid >> 2;
  const int m0   = wm * 32;
  const int n0   = wn * 64;
  const int rowBase = blockIdx.y * BM;
  const int colBase = blockIdx.x * BN;
  const int nl = lane & 15;
  const int kh = lane >> 4;

  v8f acc[2][4];
  const v8f vzero = {0.f, 0.f, 0.f, 0.f, 0.f, 0.f, 0.f, 0.f};
#pragma unroll
  for (int i = 0; i < 2; ++i)
#pragma unroll
    for (int j = 0; j < 4; ++j) acc[i][j] = vzero;

  // stage 128x32 bf16 tile pair: 512 b128 slots / 256 threads = 2 each side
  auto stage = [&](int buf, int k0) {
#pragma unroll
    for (int i = 0; i < 2; ++i) {
      int q  = tid + 256 * i;          // 0..511
      int r  = q >> 2;                 // tile row 0..127
      int c8 = q & 3;                  // 8-bf16 chunk within 32-wide row
      cp_b128(Xb + (size_t)(rowBase + r) * DIM + k0 + c8 * 8,
              &lsA[buf][r * LDH + c8 * 8]);
      cp_b128(Wb + (size_t)(colBase + r) * DIM + k0 + c8 * 8,
              &lsB[buf][r * LDH + c8 * 8]);
    }
  };

  constexpr int NK = DIM / BK;         // 64 K-steps
  stage(0, 0);                         // prologue: 4 async ops in flight

  for (int it = 0; it < NK; ++it) {
    const int k0  = it * BK;
    const int cur = it & 1;
    if (it + 1 < NK) {
      stage(cur ^ 1, k0 + BK);         // 4 more async ops (other buffer)
      wait_async<4>();                 // buffer `cur` complete (in-order)
    } else {
      wait_async<0>();
    }
    __syncthreads();

    // ---- fragments straight from bf16 LDS (ISA 16-bit A/B layouts) ----
    Frag a[2], b[4];
#pragma unroll
    for (int i = 0; i < 2; ++i) {
      const unsigned short* pa = &lsA[cur][(m0 + i * 16 + nl) * LDH];
      a[i].q[0] = *(const v4u*)(pa + 8 * kh);        // K = 8*kh .. +7
      a[i].q[1] = *(const v4u*)(pa + 16 + 8 * kh);   // K = 16+8*kh .. +7
    }
#pragma unroll
    for (int j = 0; j < 4; ++j) {      // B[k,n] = W[n,k]: row n, K contiguous
      const unsigned short* pb = &lsB[cur][(n0 + j * 16 + nl) * LDH + 8 * kh];
      b[j].q[0] = *(const v4u*)pb;                   // K = 8*kh .. +7, 16+8*kh .. +7
      b[j].q[1] = *(const v4u*)(pb + 16);
    }

    // ---- 8 x v_wmma_f32_16x16x32_bf16 ----
#pragma unroll
    for (int i = 0; i < 2; ++i)
#pragma unroll
      for (int j = 0; j < 4; ++j)
        acc[i][j] = __builtin_amdgcn_wmma_f32_16x16x32_bf16(
            false, a[i].v, false, b[j].v, (short)0, acc[i][j], false, false);

    __syncthreads();
  }

  // ---- epilogue: C layout lanes 0-15 -> M=r, lanes 16-31 -> M=8+r ----
#pragma unroll
  for (int j = 0; j < 4; ++j) {
    int col  = colBase + n0 + j * 16 + nl;
    float bv = bias[col];
#pragma unroll
    for (int i = 0; i < 2; ++i) {
#pragma unroll
      for (int r = 0; r < 8; ++r) {
        int row = rowBase + m0 + i * 16 + 8 * kh + r;
        Y[(size_t)row * DIM + col] = acc[i][j][r] + bv;
      }
    }
  }
}

// ---------------------------------------------------------------------------
// Kernel 1' (fallback, small d_ws): f32 in global, pack-once-at-stage-time,
// single-buffer LDS (bf16), same fragment layout + WMMAs.
// ---------------------------------------------------------------------------
__global__ __launch_bounds__(256)
void gemm_bias_sync_kernel(const float* __restrict__ X, const float* __restrict__ Wm,
                           const float* __restrict__ bias, float* __restrict__ Y) {
  __shared__ unsigned short lsA[BM * LDH];
  __shared__ unsigned short lsB[BN * LDH];

  const int tid  = threadIdx.x;
  const int lane = tid & 31;
  const int wid  = tid >> 5;
  const int wm   = wid & 3;
  const int wn   = wid >> 2;
  const int m0   = wm * 32;
  const int n0   = wn * 64;
  const int rowBase = blockIdx.y * BM;
  const int colBase = blockIdx.x * BN;
  const int nl = lane & 15;
  const int kh = lane >> 4;

  v8f acc[2][4];
  const v8f vzero = {0.f, 0.f, 0.f, 0.f, 0.f, 0.f, 0.f, 0.f};
#pragma unroll
  for (int i = 0; i < 2; ++i)
#pragma unroll
    for (int j = 0; j < 4; ++j) acc[i][j] = vzero;

  for (int k0 = 0; k0 < DIM; k0 += BK) {
#pragma unroll
    for (int i = 0; i < 4; ++i) {      // stage + convert once per element
      int q  = tid + 256 * i;          // 0..1023 float4 slots
      int r  = q >> 3;
      int c4 = q & 7;
      v4f xa = *(const v4f*)(X  + (size_t)(rowBase + r) * DIM + k0 + c4 * 4);
      *(unsigned*)&lsA[r * LDH + c4 * 4]     = pack_bf16x2(xa.x, xa.y);
      *(unsigned*)&lsA[r * LDH + c4 * 4 + 2] = pack_bf16x2(xa.z, xa.w);
      v4f wb = *(const v4f*)(Wm + (size_t)(colBase + r) * DIM + k0 + c4 * 4);
      *(unsigned*)&lsB[r * LDH + c4 * 4]     = pack_bf16x2(wb.x, wb.y);
      *(unsigned*)&lsB[r * LDH + c4 * 4 + 2] = pack_bf16x2(wb.z, wb.w);
    }
    __syncthreads();

    Frag a[2], b[4];
#pragma unroll
    for (int i = 0; i < 2; ++i) {
      const unsigned short* pa = &lsA[(m0 + i * 16 + nl) * LDH];
      a[i].q[0] = *(const v4u*)(pa + 8 * kh);
      a[i].q[1] = *(const v4u*)(pa + 16 + 8 * kh);
    }
#pragma unroll
    for (int j = 0; j < 4; ++j) {
      const unsigned short* pb = &lsB[(n0 + j * 16 + nl) * LDH + 8 * kh];
      b[j].q[0] = *(const v4u*)pb;
      b[j].q[1] = *(const v4u*)(pb + 16);
    }
#pragma unroll
    for (int i = 0; i < 2; ++i)
#pragma unroll
      for (int j = 0; j < 4; ++j)
        acc[i][j] = __builtin_amdgcn_wmma_f32_16x16x32_bf16(
            false, a[i].v, false, b[j].v, (short)0, acc[i][j], false, false);
    __syncthreads();
  }

#pragma unroll
  for (int j = 0; j < 4; ++j) {
    int col  = colBase + n0 + j * 16 + nl;
    float bv = bias[col];
#pragma unroll
    for (int i = 0; i < 2; ++i) {
#pragma unroll
      for (int r = 0; r < 8; ++r) {
        int row = rowBase + m0 + i * 16 + 8 * kh + r;
        Y[(size_t)row * DIM + col] = acc[i][j][r] + bv;
      }
    }
  }
}

// ---------------------------------------------------------------------------
// Kernel 2: per-row mean / rsqrt(var+eps) over D=2048  ->  4096 float2
// ---------------------------------------------------------------------------
__global__ __launch_bounds__(256)
void ln_stats_kernel(const float* __restrict__ Y, float2* __restrict__ st) {
  __shared__ float sh[256], sh2[256];
  const int row = blockIdx.x;
  const float* p = Y + (size_t)row * DIM;
  float s = 0.f, s2 = 0.f;
  for (int i = threadIdx.x; i < DIM; i += 256) {
    float v = p[i];
    s += v; s2 += v * v;
  }
  sh[threadIdx.x] = s; sh2[threadIdx.x] = s2;
  __syncthreads();
  for (int o = 128; o > 0; o >>= 1) {
    if (threadIdx.x < o) {
      sh[threadIdx.x]  += sh[threadIdx.x + o];
      sh2[threadIdx.x] += sh2[threadIdx.x + o];
    }
    __syncthreads();
  }
  if (threadIdx.x == 0) {
    float mu  = sh[0]  * (1.0f / DIM);
    float var = sh2[0] * (1.0f / DIM) - mu * mu;
    st[row] = make_float2(mu, rsqrtf(var + 1e-5f));
  }
}

// ---------------------------------------------------------------------------
// Kernel 3: LN apply + residual + 32-step LIF scan (in-place on d_out).
// tau=2, v_th=1, v_reset=0, decay_input => h = (v+z)/2 ; hard reset.
// ---------------------------------------------------------------------------
__global__ __launch_bounds__(256)
void lif_kernel(float* __restrict__ Y, const float* __restrict__ X,
                const float2* __restrict__ st,
                const float* __restrict__ lnw, const float* __restrict__ lnb) {
  const int idx = blockIdx.x * 256 + threadIdx.x;  // over B*D = 262144
  const int d   = idx & (DIM - 1);
  const int b   = idx >> 11;
  const float w  = lnw[d];
  const float bb = lnb[d];
  float v = 0.f;
#pragma unroll 4
  for (int t = 0; t < T_STEPS; ++t) {
    size_t off = (size_t)t * (BATCH * DIM) + idx;
    float2 s = st[t * BATCH + b];
    float z  = (Y[off] - s.x) * s.y * w + bb + X[off];
    float h  = 0.5f * (v + z);
    float spike = (h >= 1.0f) ? 1.0f : 0.0f;
    v = h - h * spike;
    Y[off] = spike;
  }
}

// ---------------------------------------------------------------------------
extern "C" void kernel_launch(void* const* d_in, const int* in_sizes, int n_in,
                              void* d_out, int out_size, void* d_ws, size_t ws_size,
                              hipStream_t stream) {
  (void)in_sizes; (void)n_in; (void)out_size;
  const float* X    = (const float*)d_in[0];   // [T,B,D]
  const float* Wm   = (const float*)d_in[1];   // [D,D]
  const float* bias = (const float*)d_in[2];   // [D]
  const float* lnw  = (const float*)d_in[3];   // [D]
  const float* lnb  = (const float*)d_in[4];   // [D]
  float* Y = (float*)d_out;                    // reused: GEMM result, then spikes

  const size_t xbBytes   = (size_t)MTOT * DIM * 2;   // 16 MB
  const size_t wbBytes   = (size_t)DIM  * DIM * 2;   //  8 MB
  const size_t statsOff  = xbBytes + wbBytes;
  const size_t needFast  = statsOff + (size_t)MTOT * sizeof(float2);

  dim3 g1(DIM / BN, MTOT / BM);                // (16, 32)

  if (ws_size >= needFast) {
    unsigned short* Xb = (unsigned short*)d_ws;
    unsigned short* Wb = (unsigned short*)((char*)d_ws + xbBytes);
    float2* st = (float2*)((char*)d_ws + statsOff);
    const int nx8 = MTOT * DIM / 8;            // 1,048,576
    const int nw8 = DIM * DIM / 8;             //   524,288
    f32_to_bf16_kernel<<<(nx8 + 255) / 256, 256, 0, stream>>>(X,  Xb, nx8);
    f32_to_bf16_kernel<<<(nw8 + 255) / 256, 256, 0, stream>>>(Wm, Wb, nw8);
    gemm_bias_bf16_kernel<<<g1, 256, 0, stream>>>(Xb, Wb, bias, Y);
    ln_stats_kernel<<<MTOT, 256, 0, stream>>>(Y, st);
    lif_kernel<<<(BATCH * DIM) / 256, 256, 0, stream>>>(Y, X, st, lnw, lnb);
  } else {
    float2* st = (float2*)d_ws;                // 32 KB
    gemm_bias_sync_kernel<<<g1, 256, 0, stream>>>(X, Wm, bias, Y);
    ln_stats_kernel<<<MTOT, 256, 0, stream>>>(Y, st);
    lif_kernel<<<(BATCH * DIM) / 256, 256, 0, stream>>>(Y, X, st, lnw, lnb);
  }
}
